// Architecture_13829794693845
// MI455X (gfx1250) — matrix operand order repaired and checked by reading, not compiled, not run on hardware
//
#include <hip/hip_runtime.h>

// ---------------------------------------------------------------------------
// AKT-style transformer forward for MI455X (gfx1250, wave32, WMMA).
// GEMMs in bf16 with f32 accumulation via v_wmma_f32_16x16x32_bf16,
// double-buffered LDS tiles fed by GLOBAL_LOAD_ASYNC_TO_LDS_B128 when the
// toolchain exposes the builtin (ASYNCcnt pipeline), sync fallback otherwise.
// Softmax / decay / LayerNorm in fp32.
// ---------------------------------------------------------------------------

typedef unsigned short u16;
typedef __attribute__((ext_vector_type(16))) __bf16 v16bf;
typedef __attribute__((ext_vector_type(8)))  __bf16 v8bf;
typedef __attribute__((ext_vector_type(8)))  float  v8f;
typedef int v4i __attribute__((vector_size(16)));    // matches ROCm builtin param type

enum : int { B = 32, S = 512, D = 256, H = 8, DK = 32, DFF = 1024, L = 6 };

#define DEV __device__ __forceinline__

#if defined(__AMDGCN__) && __has_builtin(__builtin_amdgcn_global_load_async_to_lds_b128) && \
    __has_builtin(__builtin_amdgcn_s_wait_asynccnt)
#define HAS_ASYNC 1
// global (addrspace 1) int4* ; LDS (addrspace 3) int4*
#define GPTR(p) ((__attribute__((address_space(1))) v4i*)(p))
#define LPTR(p) ((__attribute__((address_space(3))) v4i*)(p))
#else
#define HAS_ASYNC 0
#endif

DEV u16 f32_to_bf16(float f) {
  union { float f; unsigned u; } c; c.f = f;
  unsigned u = c.u;
  unsigned r = u + 0x7FFFu + ((u >> 16) & 1u);
  if ((u & 0x7F800000u) == 0x7F800000u) r = u;   // inf/nan: truncate
  return (u16)(r >> 16);
}

DEV v8bf zero8() {
  union { v8bf v; u16 s[8]; } z;
#pragma unroll
  for (int i = 0; i < 8; ++i) z.s[i] = 0;
  return z.v;
}

// ------------------------------ elementwise --------------------------------

__global__ void k_f32_to_bf16(const float* __restrict__ in, u16* __restrict__ out, int n) {
  int i = blockIdx.x * blockDim.x + threadIdx.x;
  if (i < n) out[i] = f32_to_bf16(in[i]);
}

// W [K,N] f32 (layer = blockIdx.z) -> Wt [N,K] bf16
__global__ void k_transpose_w(const float* __restrict__ W, u16* __restrict__ Wt, int K, int N) {
  long base = (long)blockIdx.z * K * N;
  int i = blockIdx.x * blockDim.x + threadIdx.x;
  if (i >= K * N) return;
  int n = i / K, k = i - n * K;
  Wt[base + i] = f32_to_bf16(W[base + (long)k * N + n]);
}

// proj [B*S, D] f32 -> Qh [B,H,S,DK] bf16
__global__ void k_split_heads(const float* __restrict__ proj, u16* __restrict__ out) {
  int i = blockIdx.x * blockDim.x + threadIdx.x;
  if (i >= B * S * D) return;
  int dk = i % DK, h = (i / DK) % H;
  int row = i / D;               // b*S + s
  int s = row % S, b = row / S;
  out[(((long)(b * H + h) * S + s) * DK) + dk] = f32_to_bf16(proj[i]);
}

// proj [B*S, D] f32 -> Vt [B,H,DK,S] bf16 (transposed for the PV GEMM)
__global__ void k_split_heads_T(const float* __restrict__ proj, u16* __restrict__ out) {
  int i = blockIdx.x * blockDim.x + threadIdx.x;
  if (i >= B * S * D) return;
  int dk = i % DK, h = (i / DK) % H;
  int row = i / D;
  int s = row % S, b = row / S;
  out[(((long)(b * H + h) * DK + dk) * S) + s] = f32_to_bf16(proj[i]);
}

// Oh [B,H,S,DK] f32 -> out [B*S, D] bf16
__global__ void k_merge_heads(const float* __restrict__ Oh, u16* __restrict__ out) {
  int i = blockIdx.x * blockDim.x + threadIdx.x;
  if (i >= B * S * D) return;
  int dk = i % DK, h = (i / DK) % H;
  int row = i / D;
  int s = row % S, b = row / S;
  out[i] = f32_to_bf16(Oh[(((long)(b * H + h) * S + s) * DK) + dk]);
}

// ------------------------------- WMMA GEMM ---------------------------------
// C[z][M,N] = alpha * A[z][M,K] * Bt[z][N,K]^T + bias[N]   (opt. relu)
// A, Bt bf16 row-major; Cf fp32 and/or Cb bf16, ldc = N.
// Block: 256 threads = 8 waves; tile 128x64; wave 32x32 (2x2 16x16 frags);
// K step 32; double-buffered LDS (async-to-LDS pipeline when available).
// GUARD=false requires M%128==0, N%64==0, K%32==0 (true for all but PV GEMM).
template <bool GUARD>
__global__ __launch_bounds__(256) void k_gemm_t(
    const u16* __restrict__ A, const u16* __restrict__ Bt,
    const float* __restrict__ bias,
    float* __restrict__ Cf, u16* __restrict__ Cb,
    int M, int N, int K,
    long strideA, long strideB, long strideC,
    float alpha, int relu)
{
  __shared__ __align__(32) u16 As[2][128 * 32];
  __shared__ __align__(32) u16 Bs[2][64 * 32];

  const int tid  = threadIdx.x;
  const int lane = tid & 31;
  const int wave = tid >> 5;
  const int wm = (wave >> 1) * 32;     // 0,32,64,96
  const int wn = (wave & 1) * 32;      // 0,32
  const int bm = blockIdx.y * 128;
  const int bn = blockIdx.x * 64;
  const long z = blockIdx.z;
  A  += z * strideA;
  Bt += z * strideB;

  // per-thread tile-load coordinates (16B per instruction)
  const int lr = tid >> 2;             // 0..63
  const int lc = (tid & 3) * 8;        // 0,8,16,24

  auto issue_loads = [&](int k0, int bufi) {
    if constexpr (!GUARD) {
#if HAS_ASYNC
      const u16* ga0 = A  + (long)(bm + lr) * K + k0 + lc;
      const u16* ga1 = A  + (long)(bm + lr + 64) * K + k0 + lc;
      const u16* gb  = Bt + (long)(bn + lr) * K + k0 + lc;
      __builtin_amdgcn_global_load_async_to_lds_b128(
          GPTR(ga0), LPTR(&As[bufi][lr * 32 + lc]), 0, 0);
      __builtin_amdgcn_global_load_async_to_lds_b128(
          GPTR(ga1), LPTR(&As[bufi][(lr + 64) * 32 + lc]), 0, 0);
      __builtin_amdgcn_global_load_async_to_lds_b128(
          GPTR(gb),  LPTR(&Bs[bufi][lr * 32 + lc]), 0, 0);
#else
      *(v8bf*)(void*)(&As[bufi][lr * 32 + lc]) =
          *(const v8bf*)(const void*)(A + (long)(bm + lr) * K + k0 + lc);
      *(v8bf*)(void*)(&As[bufi][(lr + 64) * 32 + lc]) =
          *(const v8bf*)(const void*)(A + (long)(bm + lr + 64) * K + k0 + lc);
      *(v8bf*)(void*)(&Bs[bufi][lr * 32 + lc]) =
          *(const v8bf*)(const void*)(Bt + (long)(bn + lr) * K + k0 + lc);
#endif
    } else {
      v8bf va0 = zero8(), va1 = zero8(), vb = zero8();
      if (bm + lr < M)      va0 = *(const v8bf*)(const void*)(A + (long)(bm + lr) * K + k0 + lc);
      if (bm + lr + 64 < M) va1 = *(const v8bf*)(const void*)(A + (long)(bm + lr + 64) * K + k0 + lc);
      if (bn + lr < N)      vb  = *(const v8bf*)(const void*)(Bt + (long)(bn + lr) * K + k0 + lc);
      *(v8bf*)(void*)(&As[bufi][lr * 32 + lc])        = va0;
      *(v8bf*)(void*)(&As[bufi][(lr + 64) * 32 + lc]) = va1;
      *(v8bf*)(void*)(&Bs[bufi][lr * 32 + lc])        = vb;
    }
  };

  v8f acc[2][2];
#pragma unroll
  for (int i = 0; i < 2; ++i)
#pragma unroll
    for (int j = 0; j < 2; ++j)
#pragma unroll
      for (int r = 0; r < 8; ++r) acc[i][j][r] = 0.0f;

  const int hl  = lane >> 4;     // which K-half this lane holds
  const int l16 = lane & 15;

  issue_loads(0, 0);
  int buf = 0;
  for (int k0 = 0; k0 < K; k0 += 32) {
    const bool more = (k0 + 32 < K);
    if (more) issue_loads(k0 + 32, buf ^ 1);
#if HAS_ASYNC
    if constexpr (!GUARD) {
      if (more) __builtin_amdgcn_s_wait_asynccnt(3);   // own loads for 'buf' done
      else      __builtin_amdgcn_s_wait_asynccnt(0);
    }
#endif
    __syncthreads();             // all waves' loads for 'buf' visible

#pragma unroll
    for (int fi = 0; fi < 2; ++fi) {
      int mrow = wm + fi * 16 + l16;
      v16bf av = *(const v16bf*)(const void*)(&As[buf][mrow * 32 + hl * 16]);     // K contiguous
#pragma unroll
      for (int fj = 0; fj < 2; ++fj) {
        int nrow = wn + fj * 16 + l16;
        v16bf bv = *(const v16bf*)(const void*)(&Bs[buf][nrow * 32 + hl * 16]);   // K contiguous
        acc[fi][fj] = __builtin_amdgcn_wmma_f32_16x16x32_bf16(
            false, av, false, bv, (short)0, acc[fi][fj], false, false);
      }
    }
    __syncthreads();             // everyone done reading 'buf' before reuse
    buf ^= 1;
  }

#pragma unroll
  for (int fi = 0; fi < 2; ++fi) {
#pragma unroll
    for (int fj = 0; fj < 2; ++fj) {
      int col = bn + wn + fj * 16 + l16;
      if (GUARD && col >= N) continue;
      float bvv = bias ? bias[col] : 0.0f;
#pragma unroll
      for (int r = 0; r < 8; ++r) {
        int row = bm + wm + fi * 16 + hl * 8 + r;
        if (GUARD && row >= M) continue;
        float v = acc[fi][fj][r] * alpha + bvv;
        if (relu) v = fmaxf(v, 0.0f);
        long idx = z * strideC + (long)row * N + col;
        if (Cf) Cf[idx] = v;
        if (Cb) Cb[idx] = f32_to_bf16(v);
      }
    }
  }
}

// ----------------------- attention decay + softmax -------------------------
// One block (512 threads) per (row i, head h); scores chunk is [H,S,S] for one b.
__launch_bounds__(512)
__global__ void k_attn_decay(const float* __restrict__ scores, u16* __restrict__ probs,
                             const float* __restrict__ gammas, int incl_diag, int zero_pad)
{
  const int i = blockIdx.x;
  const int h = blockIdx.y;
  const int j = threadIdx.x;
  __shared__ float buf[S];

  const long base = ((long)h * S + i) * S;
  const float s   = scores[base + j];
  const bool  m   = incl_diag ? (j <= i) : (j < i);
  const float NEGV = -1e32f;

  // softmax #1
  float x1 = m ? s : NEGV;
  buf[j] = x1; __syncthreads();
  for (int off = 256; off > 0; off >>= 1) { if (j < off) buf[j] = fmaxf(buf[j], buf[j + off]); __syncthreads(); }
  float mx = buf[0]; __syncthreads();
  float e = __expf(x1 - mx);
  buf[j] = e; __syncthreads();
  for (int off = 256; off > 0; off >>= 1) { if (j < off) buf[j] += buf[j + off]; __syncthreads(); }
  float sum = buf[0]; __syncthreads();
  float sc = m ? (e / sum) : 0.0f;

  // inclusive prefix sum of sc
  buf[j] = sc; __syncthreads();
  for (int off = 1; off < S; off <<= 1) {
    float v = (j >= off) ? buf[j - off] : 0.0f;
    __syncthreads();
    buf[j] += v;
    __syncthreads();
  }
  float cum = buf[j];
  float tot = buf[S - 1];
  __syncthreads();

  float pos  = fabsf((float)i - (float)j);
  float dist = sqrtf(fmaxf((tot - cum) * pos, 0.0f));
  float gm   = gammas[h];
  float sp   = (gm > 20.0f) ? gm : log1pf(__expf(gm));   // softplus
  float eff  = __expf(dist * -sp);
  eff = fminf(fmaxf(eff, 1e-5f), 1e5f);

  // softmax #2 on decayed scores
  float x2 = m ? (s * eff) : NEGV;
  buf[j] = x2; __syncthreads();
  for (int off = 256; off > 0; off >>= 1) { if (j < off) buf[j] = fmaxf(buf[j], buf[j + off]); __syncthreads(); }
  float mx2 = buf[0]; __syncthreads();
  float e2 = __expf(x2 - mx2);
  buf[j] = e2; __syncthreads();
  for (int off = 256; off > 0; off >>= 1) { if (j < off) buf[j] += buf[j + off]; __syncthreads(); }
  float sum2 = buf[0];

  float a = e2 / sum2;
  if (zero_pad && i == 0) a = 0.0f;
  probs[base + j] = f32_to_bf16(a);
}

// --------------------------- residual + LayerNorm --------------------------
__launch_bounds__(256)
__global__ void k_add_ln(float* __restrict__ res, const float* __restrict__ o,
                         const float* __restrict__ sg, const float* __restrict__ bg)
{
  const int r = blockIdx.x, t = threadIdx.x;
  __shared__ float buf[D];
  const long idx = (long)r * D + t;
  float v = res[idx] + o[idx];
  buf[t] = v; __syncthreads();
  for (int off = 128; off > 0; off >>= 1) { if (t < off) buf[t] += buf[t + off]; __syncthreads(); }
  float mu = buf[0] * (1.0f / D); __syncthreads();
  float d = v - mu;
  buf[t] = d * d; __syncthreads();
  for (int off = 128; off > 0; off >>= 1) { if (t < off) buf[t] += buf[t + off]; __syncthreads(); }
  float var = buf[0] * (1.0f / D);
  res[idx] = d * rsqrtf(var + 1e-5f) * sg[t] + bg[t];
}

// ------------------------------- host driver -------------------------------

extern "C" void kernel_launch(void* const* d_in, const int* in_sizes, int n_in,
                              void* d_out, int out_size, void* d_ws, size_t ws_size,
                              hipStream_t stream)
{
  (void)in_sizes; (void)n_in; (void)out_size; (void)ws_size;
  const int M = B * S;

  const float* q_embed  = (const float*)d_in[0];
  const float* qa_embed = (const float*)d_in[1];
  const float* Wk = (const float*)d_in[2];
  const float* bk = (const float*)d_in[3];
  const float* Wv = (const float*)d_in[4];
  const float* bv = (const float*)d_in[5];
  const float* Wo = (const float*)d_in[6];
  const float* bo = (const float*)d_in[7];
  const float* gammas = (const float*)d_in[8];
  const float* ln1_s = (const float*)d_in[9];
  const float* ln1_b = (const float*)d_in[10];
  const float* W1 = (const float*)d_in[11];
  const float* b1 = (const float*)d_in[12];
  const float* W2 = (const float*)d_in[13];
  const float* b2 = (const float*)d_in[14];
  const float* ln2_s = (const float*)d_in[15];
  const float* ln2_b = (const float*)d_in[16];

  char* wsp = (char*)d_ws;
  auto alloc = [&](size_t bytes) -> void* {
    void* p = (void*)wsp;
    wsp += (bytes + 255) & ~(size_t)255;
    return p;
  };

  u16*   WkT      = (u16*)  alloc((size_t)L * D * D * 2);
  u16*   WvT      = (u16*)  alloc((size_t)L * D * D * 2);
  u16*   WoT      = (u16*)  alloc((size_t)L * D * D * 2);
  u16*   W1T      = (u16*)  alloc((size_t)L * D * DFF * 2);
  u16*   W2T      = (u16*)  alloc((size_t)L * DFF * D * 2);
  float* x_f32    = (float*)alloc((size_t)M * D * 4);
  float* y_f32    = (float*)alloc((size_t)M * D * 4);
  u16*   act_bf16 = (u16*)  alloc((size_t)M * D * 2);
  float* proj_f32 = (float*)alloc((size_t)M * D * 4);
  u16*   Qh       = (u16*)  alloc((size_t)B * H * S * DK * 2);
  u16*   Vt       = (u16*)  alloc((size_t)B * H * DK * S * 2);
  float* scores   = (float*)alloc((size_t)H * S * S * 4);
  u16*   probs    = (u16*)  alloc((size_t)H * S * S * 2);
  float* Oh       = (float*)alloc((size_t)B * H * S * DK * 4);
  u16*   attnout  = (u16*)  alloc((size_t)M * D * 2);
  float* o_f32    = (float*)alloc((size_t)M * D * 4);
  u16*   f1_bf16  = (u16*)  alloc((size_t)M * DFF * 2);

  // Pre-transpose+convert all weights to bf16 [N,K] once.
  {
    int n = D * D, n1 = D * DFF;
    k_transpose_w<<<dim3((n  + 255) / 256, 1, L), 256, 0, stream>>>(Wk, WkT, D, D);
    k_transpose_w<<<dim3((n  + 255) / 256, 1, L), 256, 0, stream>>>(Wv, WvT, D, D);
    k_transpose_w<<<dim3((n  + 255) / 256, 1, L), 256, 0, stream>>>(Wo, WoT, D, D);
    k_transpose_w<<<dim3((n1 + 255) / 256, 1, L), 256, 0, stream>>>(W1, W1T, D, DFF);
    k_transpose_w<<<dim3((n1 + 255) / 256, 1, L), 256, 0, stream>>>(W2, W2T, DFF, D);
  }

  (void)hipMemcpyAsync(x_f32, q_embed,  (size_t)M * D * 4, hipMemcpyDeviceToDevice, stream);
  (void)hipMemcpyAsync(y_f32, qa_embed, (size_t)M * D * 4, hipMemcpyDeviceToDevice, stream);

  const int eg = (M * D + 255) / 256;
  const float inv_sqrt_dk = 0.17677669529663687f;   // 1/sqrt(32)

  auto run_layer = [&](int i, int mask_flag, float* qres, const float* vin, bool apply_pos) {
    dim3 gP(D / 64, M / 128, 1);
    // q/k projection (kq_same, and q==k input in every layer of this net)
    k_f32_to_bf16<<<eg, 256, 0, stream>>>(qres, act_bf16, M * D);
    k_gemm_t<false><<<gP, 256, 0, stream>>>(act_bf16, WkT + (long)i * D * D, bk + (long)i * D,
                                            proj_f32, nullptr, M, D, D, 0, 0, 0, 1.0f, 0);
    k_split_heads<<<eg, 256, 0, stream>>>(proj_f32, Qh);

    // v projection (stored transposed per head)
    k_f32_to_bf16<<<eg, 256, 0, stream>>>(vin, act_bf16, M * D);
    k_gemm_t<false><<<gP, 256, 0, stream>>>(act_bf16, WvT + (long)i * D * D, bv + (long)i * D,
                                            proj_f32, nullptr, M, D, D, 0, 0, 0, 1.0f, 0);
    k_split_heads_T<<<eg, 256, 0, stream>>>(proj_f32, Vt);

    // attention, chunked per batch index (score scratch = H*S*S fp32)
    for (int b = 0; b < B; ++b) {
      const u16* Qb = Qh + (long)b * H * S * DK;
      const u16* Vb = Vt + (long)b * H * DK * S;
      k_gemm_t<false><<<dim3(S / 64, S / 128, H), 256, 0, stream>>>(
          Qb, Qb, nullptr, scores, nullptr, S, S, DK,
          (long)S * DK, (long)S * DK, (long)S * S, inv_sqrt_dk, 0);
      k_attn_decay<<<dim3(S, H), 512, 0, stream>>>(
          scores, probs, gammas + (long)i * H, mask_flag, mask_flag == 0 ? 1 : 0);
      k_gemm_t<true><<<dim3(1, S / 128, H), 256, 0, stream>>>(
          probs, Vb, nullptr, Oh + (long)b * H * S * DK, nullptr, S, DK, S,
          (long)S * S, (long)DK * S, (long)S * DK, 1.0f, 0);
    }
    k_merge_heads<<<eg, 256, 0, stream>>>(Oh, attnout);

    // output projection + residual LN
    k_gemm_t<false><<<gP, 256, 0, stream>>>(attnout, WoT + (long)i * D * D, bo + (long)i * D,
                                            o_f32, nullptr, M, D, D, 0, 0, 0, 1.0f, 0);
    k_add_ln<<<M, 256, 0, stream>>>(qres, o_f32, ln1_s + (long)i * D, ln1_b + (long)i * D);

    if (apply_pos) {
      k_f32_to_bf16<<<eg, 256, 0, stream>>>(qres, act_bf16, M * D);
      k_gemm_t<false><<<dim3(DFF / 64, M / 128, 1), 256, 0, stream>>>(
          act_bf16, W1T + (long)i * D * DFF, b1 + (long)i * DFF,
          nullptr, f1_bf16, M, DFF, D, 0, 0, 0, 1.0f, 1);
      k_gemm_t<false><<<dim3(D / 64, M / 128, 1), 256, 0, stream>>>(
          f1_bf16, W2T + (long)i * DFF * D, b2 + (long)i * D,
          o_f32, nullptr, M, D, DFF, 0, 0, 0, 1.0f, 0);
      k_add_ln<<<M, 256, 0, stream>>>(qres, o_f32, ln2_s + (long)i * D, ln2_b + (long)i * D);
    }
  };

  // blocks_1: two self-attn layers on y (incl-diag mask, with FFN)
  run_layer(0, 1, y_f32, y_f32, true);
  run_layer(1, 1, y_f32, y_f32, true);
  // blocks_2: alternate on x
  run_layer(2, 1, x_f32, x_f32, false);
  run_layer(3, 0, x_f32, y_f32, true);
  run_layer(4, 1, x_f32, x_f32, false);
  run_layer(5, 0, x_f32, y_f32, true);

  (void)hipMemcpyAsync(d_out, x_f32, (size_t)M * D * 4, hipMemcpyDeviceToDevice, stream);
}
